// PoplarCrossOmicsModel_50723563766184
// MI455X (gfx1250) — compile-verified, matrix operand order checked by reading, not run on hardware
//
#include <hip/hip_runtime.h>
#include <math.h>

typedef _Float16 v16h __attribute__((ext_vector_type(16)));
typedef _Float16 v8h  __attribute__((ext_vector_type(8)));
typedef float    v8f  __attribute__((ext_vector_type(8)));

#define NN   100000
#define EE   800000
#define DIN  128
#define DOUT 128
#define NH   8
#define DK   16
#define NT   3
#define NR   8

// -------------------- helpers --------------------
__device__ __forceinline__ void atomicMaxF(float* addr, float val) {
    if (val >= 0.0f) {
        atomicMax((int*)addr, __float_as_int(val));
    } else {
        atomicMin((unsigned int*)addr, __float_as_uint(val));
    }
}

__device__ __forceinline__ float wave_sum32(float v) {
    #pragma unroll
    for (int off = 16; off > 0; off >>= 1) v += __shfl_xor(v, off, 32);
    return v;
}

// -------------------- conversion kernels --------------------
__global__ void k_cvt_node(const float* __restrict__ in, _Float16* __restrict__ out, int n) {
    int i = blockIdx.x * blockDim.x + threadIdx.x;
    if (i < n) out[i] = (_Float16)in[i];
}

// Build Bt[proj][t][o][k] (f16) = W[proj][t][k][o]  (transposed for B-fragment loads)
__global__ void k_cvt_w(const float* __restrict__ Wq, const float* __restrict__ Wk,
                        const float* __restrict__ Wv, _Float16* __restrict__ Bt) {
    int idx = blockIdx.x * blockDim.x + threadIdx.x;     // 9*128*128
    if (idx >= 9 * DIN * DOUT) return;
    int k  = idx & 127;
    int o  = (idx >> 7) & 127;
    int pt = idx >> 14;          // 0..8
    int proj = pt / 3, t = pt % 3;
    const float* W = (proj == 0) ? Wq : (proj == 1) ? Wk : Wv;
    Bt[idx] = (_Float16)W[(t * DIN + k) * DOUT + o];
}

__global__ void k_init(float* __restrict__ agg, float* __restrict__ nmax,
                       float* __restrict__ nz) {
    int i = blockIdx.x * blockDim.x + threadIdx.x;
    if (i < NN * DOUT) agg[i] = 0.0f;
    if (i < NN * NH) { nmax[i] = -3.402823466e38f; nz[i] = 0.0f; }
}

// -------------------- WMMA QKV projection --------------------
// grid = (N/16, 8 col tiles, 3 projections), 1 wave32 per block.
// Computes all 3 type-variants per tile, selects per-row by node_type.
__global__ __launch_bounds__(32)
void k_qkv_wmma(const _Float16* __restrict__ Af,   // [N][128] f16
                const _Float16* __restrict__ Bt,   // [3][3][128][128] f16 (proj,t,o,k)
                const float* __restrict__ bq, const float* __restrict__ bk,
                const float* __restrict__ bv,
                const int* __restrict__ node_type,
                float* __restrict__ QKV)           // [3][N][128] f32
{
    const int nt   = blockIdx.x;
    const int ct   = blockIdx.y;
    const int proj = blockIdx.z;
    const int l  = threadIdx.x;
    const int m  = l & 15;       // row (A) / col (B) within tile
    const int kh = l >> 4;       // which K-half this lane owns

    // A fragments: 16-bit A 16x32 layout — lane(0-15): K 0..7 & 16..23; lane(16-31): K 8..15 & 24..31
    v16h a[4];
    const _Float16* arow = Af + (size_t)(nt * 16 + m) * DIN;
    #pragma unroll
    for (int kc = 0; kc < 4; ++kc) {
        const _Float16* p = arow + kc * 32 + kh * 8;
        v8h lo = *(const v8h*)p;
        v8h hi = *(const v8h*)(p + 16);
        #pragma unroll
        for (int i = 0; i < 8; ++i) { a[kc][i] = lo[i]; a[kc][8 + i] = hi[i]; }
    }

    v8f acc[3];
    #pragma unroll
    for (int t = 0; t < 3; ++t) {
        v8f c = {0.f, 0.f, 0.f, 0.f, 0.f, 0.f, 0.f, 0.f};
        const _Float16* brow = Bt + (((size_t)proj * 3 + t) * DOUT + (ct * 16 + m)) * DIN;
        #pragma unroll
        for (int kc = 0; kc < 4; ++kc) {
            const _Float16* p = brow + kc * 32 + kh * 8;
            v8h lo = *(const v8h*)p;
            v8h hi = *(const v8h*)(p + 16);
            v16h b;
            #pragma unroll
            for (int i = 0; i < 8; ++i) { b[i] = lo[i]; b[8 + i] = hi[i]; }
            c = __builtin_amdgcn_wmma_f32_16x16x32_f16(false, a[kc], false, b,
                                                       (short)0, c, false, false);
        }
        acc[t] = c;
    }

    const float* bias = (proj == 0) ? bq : (proj == 1) ? bk : bv;
    const int col = ct * 16 + m;
    #pragma unroll
    for (int i = 0; i < 8; ++i) {
        const int node = nt * 16 + i + kh * 8;   // D layout: lanes>=16 hold M=i+8
        const int t = node_type[node];
        const float v = ((t == 0) ? acc[0][i] : (t == 1) ? acc[1][i] : acc[2][i])
                        + bias[t * DOUT + col];
        QKV[((size_t)proj * NN + node) * DOUT + col] = v;
    }
}

// -------------------- edge scores + segment max --------------------
__global__ void k_scores(const float* __restrict__ QKV,
                         const int* __restrict__ eidx, const int* __restrict__ etype,
                         const int* __restrict__ esign, const float* __restrict__ edist,
                         const float* __restrict__ rel_q, const float* __restrict__ rel_k,
                         const float* __restrict__ rel_bias,
                         const float* __restrict__ sk_fixed, const float* __restrict__ sk_neut,
                         const float* __restrict__ p_alpha, const float* __restrict__ p_tau,
                         float* __restrict__ scores, float* __restrict__ nmax)
{
    int idx = blockIdx.x * blockDim.x + threadIdx.x;
    if (idx >= EE * NH) return;
    const int e = idx >> 3, h = idx & 7;
    const int src = eidx[e], dst = eidx[EE + e];
    const int et = etype[e], sg = esign[e];
    const int sidx = (sg == -1) ? 0 : (sg == 1) ? 1 : 2;

    const float* q  = QKV + ((size_t)0 * NN + dst) * DOUT + h * DK;
    const float* kk = QKV + ((size_t)1 * NN + src) * DOUT + h * DK;
    const float* rq = rel_q + (et * NH + h) * DK;
    const float* rk = rel_k + (et * NH + h) * DK;
    const float* sk = (sidx < 2) ? (sk_fixed + (sidx * NH + h) * DK) : (sk_neut + h * DK);

    float dot = 0.0f;
    #pragma unroll
    for (int d = 0; d < DK; ++d)
        dot += q[d] * rq[d] * kk[d] * rk[d] * sk[d];

    const float phi = p_alpha[0] * expf(-edist[e] / (p_tau[0] + 1e-9f));
    const float s = dot * 0.25f + rel_bias[et * NH + h] + phi;   // 1/sqrt(16)=0.25
    scores[idx] = s;
    atomicMaxF(&nmax[dst * NH + h], s);
}

// -------------------- exp + segment sum --------------------
__global__ void k_expsum(const int* __restrict__ eidx,
                         float* __restrict__ scores,
                         const float* __restrict__ nmax, float* __restrict__ nz)
{
    int idx = blockIdx.x * blockDim.x + threadIdx.x;
    if (idx >= EE * NH) return;
    const int e = idx >> 3, h = idx & 7;
    const int dst = eidx[EE + e];
    const float es = expf(scores[idx] - nmax[dst * NH + h]);
    scores[idx] = es;
    atomicAdd(&nz[dst * NH + h], es);
}

// -------------------- normalize + message scatter-add --------------------
__global__ void k_scatter(const float* __restrict__ QKV,
                          const int* __restrict__ eidx, const int* __restrict__ etype,
                          const int* __restrict__ esign,
                          const float* __restrict__ rel_v,
                          const float* __restrict__ sv_fixed, const float* __restrict__ sv_neut,
                          const float* __restrict__ scores, const float* __restrict__ nz,
                          float* __restrict__ agg)
{
    long long idx = (long long)blockIdx.x * blockDim.x + threadIdx.x;
    if (idx >= (long long)EE * DOUT) return;
    const int e = (int)(idx >> 7);
    const int c = (int)(idx & 127);     // c = h*16 + d
    const int h = c >> 4;
    const int src = eidx[e], dst = eidx[EE + e];
    const int et = etype[e], sg = esign[e];
    const int sidx = (sg == -1) ? 0 : (sg == 1) ? 1 : 2;

    const float sv = (sidx < 2) ? sv_fixed[sidx * DOUT + c] : sv_neut[c];
    const float veff = QKV[((size_t)2 * NN + src) * DOUT + c] * rel_v[et * DOUT + c] * sv;
    const float attn = scores[e * NH + h] / (nz[dst * NH + h] + 1e-9f);
    atomicAdd(&agg[(size_t)dst * DOUT + c], veff * attn);
}

// -------------------- gated residual + per-type LayerNorm --------------------
__global__ __launch_bounds__(256)
void k_ln(const float* __restrict__ inp, const int* __restrict__ node_type,
          const float* __restrict__ skip, const float* __restrict__ gamma,
          const float* __restrict__ beta, float* __restrict__ out /* agg in, result out */)
{
    const int node = blockIdx.x * 8 + (threadIdx.x >> 5);
    const int lane = threadIdx.x & 31;
    if (node >= NN) return;
    const int t = node_type[node];
    const float alpha = 1.0f / (1.0f + expf(-skip[t]));

    float x[4];
    #pragma unroll
    for (int j = 0; j < 4; ++j) {
        const int c = lane + 32 * j;
        x[j] = alpha * out[(size_t)node * DOUT + c]
             + (1.0f - alpha) * inp[(size_t)node * DOUT + c];
    }
    float s = x[0] + x[1] + x[2] + x[3];
    const float mu = wave_sum32(s) * (1.0f / DOUT);
    float v = 0.0f;
    #pragma unroll
    for (int j = 0; j < 4; ++j) { const float d = x[j] - mu; v += d * d; }
    const float var = wave_sum32(v) * (1.0f / DOUT);
    const float inv = rsqrtf(var + 1e-5f);
    #pragma unroll
    for (int j = 0; j < 4; ++j) {
        const int c = lane + 32 * j;
        out[(size_t)node * DOUT + c] =
            (x[j] - mu) * inv * gamma[t * DOUT + c] + beta[t * DOUT + c];
    }
}

// -------------------- launcher --------------------
extern "C" void kernel_launch(void* const* d_in, const int* in_sizes, int n_in,
                              void* d_out, int out_size, void* d_ws, size_t ws_size,
                              hipStream_t stream) {
    const float* node_inp  = (const float*)d_in[0];
    const int*   node_type = (const int*)  d_in[1];
    const int*   eidx      = (const int*)  d_in[2];
    const int*   etype     = (const int*)  d_in[3];
    const int*   esign     = (const int*)  d_in[4];
    const float* edist     = (const float*)d_in[5];
    const float* Wq = (const float*)d_in[6];  const float* bq = (const float*)d_in[7];
    const float* Wk = (const float*)d_in[8];  const float* bk = (const float*)d_in[9];
    const float* Wv = (const float*)d_in[10]; const float* bv = (const float*)d_in[11];
    const float* rel_q = (const float*)d_in[12];
    const float* rel_k = (const float*)d_in[13];
    const float* rel_v = (const float*)d_in[14];
    const float* rel_bias = (const float*)d_in[15];
    const float* sk_fixed = (const float*)d_in[16];
    const float* sv_fixed = (const float*)d_in[17];
    const float* sk_neut  = (const float*)d_in[18];
    const float* sv_neut  = (const float*)d_in[19];
    const float* p_alpha  = (const float*)d_in[20];
    const float* p_tau    = (const float*)d_in[21];
    const float* skip     = (const float*)d_in[22];
    const float* ln_gamma = (const float*)d_in[23];
    const float* ln_beta  = (const float*)d_in[24];
    float* out = (float*)d_out;

    // workspace carve-up (all 256B-aligned)
    char* ws = (char*)d_ws;
    _Float16* node_f16 = (_Float16*)ws;                 ws += (size_t)NN * DIN * 2;        // 25.6 MB
    _Float16* Bt       = (_Float16*)ws;                 ws += (size_t)9 * DIN * DOUT * 2;  // 288 KB
    float*    QKV      = (float*)ws;                    ws += (size_t)3 * NN * DOUT * 4;   // 153.6 MB
    float*    scores   = (float*)ws;                    ws += (size_t)EE * NH * 4;         // 25.6 MB
    float*    nmax     = (float*)ws;                    ws += (size_t)NN * NH * 4;         // 3.2 MB
    float*    nz       = (float*)ws;                    /* 3.2 MB */

    // 1) f32 -> f16 node features
    {
        int n = NN * DIN;
        k_cvt_node<<<(n + 255) / 256, 256, 0, stream>>>(node_inp, node_f16, n);
    }
    // 2) transpose + convert weights
    {
        int n = 9 * DIN * DOUT;
        k_cvt_w<<<(n + 255) / 256, 256, 0, stream>>>(Wq, Wk, Wv, Bt);
    }
    // 3) init aggregate (in d_out), segment max/sum buffers
    {
        int n = NN * DOUT;
        k_init<<<(n + 255) / 256, 256, 0, stream>>>(out, nmax, nz);
    }
    // 4) WMMA QKV projection
    {
        dim3 grid(NN / 16, DOUT / 16, 3);
        k_qkv_wmma<<<grid, 32, 0, stream>>>(node_f16, Bt, bq, bk, bv, node_type, QKV);
    }
    // 5) edge scores + segment max
    {
        int n = EE * NH;
        k_scores<<<(n + 255) / 256, 256, 0, stream>>>(QKV, eidx, etype, esign, edist,
                                                      rel_q, rel_k, rel_bias,
                                                      sk_fixed, sk_neut, p_alpha, p_tau,
                                                      scores, nmax);
    }
    // 6) exp + segment sum
    {
        int n = EE * NH;
        k_expsum<<<(n + 255) / 256, 256, 0, stream>>>(eidx, scores, nmax, nz);
    }
    // 7) normalize + message scatter
    {
        long long n = (long long)EE * DOUT;
        int blocks = (int)((n + 255) / 256);
        k_scatter<<<blocks, 256, 0, stream>>>(QKV, eidx, etype, esign, rel_v,
                                              sv_fixed, sv_neut, scores, nz, out);
    }
    // 8) gated residual + LayerNorm (in place on d_out)
    {
        int blocks = (NN + 7) / 8;
        k_ln<<<blocks, 256, 0, stream>>>(node_inp, node_type, skip, ln_gamma, ln_beta, out);
    }
}
